// CapsNet_12292196401271
// MI455X (gfx1250) — compile-verified, matrix operand order checked by reading, not run on hardware
//
#include <hip/hip_runtime.h>
#include <cstdint>
#include <cstddef>

typedef __bf16 v16bf __attribute__((ext_vector_type(16)));
typedef float  v8f   __attribute__((ext_vector_type(8)));

union AFrag { v16bf v; uint4 q[2]; };

// ---------------------------------------------------------------------------
// CDNA5 async global->LDS copy (ASYNCcnt-tracked, bypasses VGPRs).
// GVS addressing: lds_off (VGPR, b32), gl_off (VGPR, b32), base (SGPR pair).
// ---------------------------------------------------------------------------
__device__ __forceinline__ void async_copy_b128(unsigned lds_off,
                                                unsigned gl_off,
                                                const void* base) {
  asm volatile("global_load_async_to_lds_b128 %0, %1, %2"
               :
               : "v"(lds_off), "v"(gl_off), "s"(base)
               : "memory");
}

// ---------------------------------------------------------------------------
// conv1: x (64,2,8,128) NCHW, k1 (3,3,2,512) HWIO, stride1 SAME, relu
//        -> act1 bf16 NHWC [64][8][128][512]
// ---------------------------------------------------------------------------
__global__ void conv1_kernel(const float* __restrict__ x,
                             const float* __restrict__ k1,
                             const float* __restrict__ b1,
                             __bf16* __restrict__ act1) {
  unsigned idx = blockIdx.x * 256u + threadIdx.x;   // 33,554,432 total
  int c = idx & 511;
  int w = (idx >> 9) & 127;
  int h = (idx >> 16) & 7;
  int b = idx >> 19;
  float acc = b1[c];
  for (int kh = 0; kh < 3; kh++) {
    int hi = h + kh - 1;
    if (hi < 0 || hi >= 8) continue;
    for (int kw = 0; kw < 3; kw++) {
      int wi = w + kw - 1;
      if (wi < 0 || wi >= 128) continue;
      #pragma unroll
      for (int ci = 0; ci < 2; ci++) {
        acc += x[((b*2 + ci)*8 + hi)*128 + wi] *
               k1[(((kh*3 + kw)*2) + ci)*512 + c];
      }
    }
  }
  act1[idx] = (__bf16)fmaxf(acc, 0.0f);
}

// ---------------------------------------------------------------------------
// Transpose conv2 weights: k2 (3,3,512,1024) fp32 -> Wt2 bf16 [1024][4608]
// kidx = (kh*3+kw)*512 + ci  (matches implicit-im2col K ordering)
// ---------------------------------------------------------------------------
__global__ void wt2_kernel(const float* __restrict__ k2,
                           __bf16* __restrict__ Wt2) {
  size_t idx = (size_t)blockIdx.x * 256u + threadIdx.x;  // 4,718,592
  int kidx = (int)(idx % 4608u);
  int co   = (int)(idx / 4608u);
  Wt2[idx] = (__bf16)k2[(size_t)kidx * 1024 + co];
}

// ---------------------------------------------------------------------------
// conv2 as implicit-im2col WMMA GEMM with double-buffered async-LDS pipeline:
//   C[m=16384][n=1024] = A[m][k=4608] * Wt2[n][k]^T + b2[n]
//   m = b*256 + ho*64 + wo ; stride-2 SAME (pad only high side)
// block = 128x128 C tile, 256 threads = 8 waves (2x4), wave = 64x32.
// Each wave issues exactly 4 async b128 transfers per k-step (deterministic),
// so `s_wait_asynccnt 4` (in-order completion) guarantees the previous tile
// landed while the next tile's transfers stay in flight under the WMMAs.
// SAME-padding OOB chunks are loaded with clamped addresses and then patched
// to zero with ds_store after the async wait.
// ---------------------------------------------------------------------------
__global__ void __launch_bounds__(256)
conv2_gemm(const __bf16* __restrict__ act1,
           const __bf16* __restrict__ Wt2,
           const float* __restrict__ b2,
           float* __restrict__ out) {
  __shared__ __bf16 As[2][128][40];
  __shared__ __bf16 Bs[2][128][40];
  const int tid  = threadIdx.x;
  const int lane = tid & 31;
  const int wv   = tid >> 5;
  const int waveM = wv & 1;         // 2 waves along M
  const int waveN = wv >> 1;        // 4 waves along N
  const int mIn   = lane & 15;
  const int kB    = (lane < 16) ? 0 : 8;
  const int blockM0 = blockIdx.x * 128;
  const int blockN0 = blockIdx.y * 128;

  // Issue one k-step tile (A implicit-im2col + B weights) into buffer `buf`.
  auto issue_tile = [&](int k0, int buf) {
    const int khw = k0 >> 9;                 // step fully inside one (kh,kw)
    const int kh = khw / 3, kw = khw % 3;
    const int ci0 = k0 & 511;
    #pragma unroll
    for (int it = 0; it < 2; ++it) {         // A: 2 chunks / thread
      int i = tid + it * 256;
      int row = i >> 2, c8 = (i & 3) * 8;
      int m = blockM0 + row;
      int b = m >> 8, rem = m & 255;
      int ho = rem >> 6, wo = rem & 63;
      int hi = 2 * ho + kh, wi = 2 * wo + kw;
      int hic = (hi < 8) ? hi : 7;           // clamp; patched to 0 later
      int wic = (wi < 128) ? wi : 127;
      unsigned goff = ((unsigned)((b*8 + hic)*128 + wic) * 512u
                       + (unsigned)(ci0 + c8)) * 2u;
      unsigned loff = (unsigned)(size_t)(&As[buf][row][c8]);
      async_copy_b128(loff, goff, act1);
    }
    #pragma unroll
    for (int it = 0; it < 2; ++it) {         // B: 2 chunks / thread
      int i = tid + it * 256;
      int row = i >> 2, c8 = (i & 3) * 8;
      unsigned goff = ((unsigned)(blockN0 + row) * 4608u
                       + (unsigned)(k0 + c8)) * 2u;
      unsigned loff = (unsigned)(size_t)(&Bs[buf][row][c8]);
      async_copy_b128(loff, goff, Wt2);
    }
  };

  // Zero-patch SAME-padding chunks of tile k0 in buffer `buf`.
  auto zero_patch = [&](int k0, int buf) {
    const int khw = k0 >> 9;
    const int kh = khw / 3, kw = khw % 3;
    #pragma unroll
    for (int it = 0; it < 2; ++it) {
      int i = tid + it * 256;
      int row = i >> 2, c8 = (i & 3) * 8;
      int m = blockM0 + row;
      int rem = m & 255;
      int ho = rem >> 6, wo = rem & 63;
      int hi = 2 * ho + kh, wi = 2 * wo + kw;
      if (hi >= 8 || wi >= 128)
        *(uint4*)(&As[buf][row][c8]) = make_uint4(0u, 0u, 0u, 0u);
    }
  };

  v8f acc[4][2] = {};
  issue_tile(0, 0);

  for (int k0 = 0; k0 < 4608; k0 += 32) {
    const int cur = (k0 >> 5) & 1;
    if (k0 + 32 < 4608) {
      issue_tile(k0 + 32, cur ^ 1);
      asm volatile("s_wait_asynccnt 0x4" ::: "memory");  // tile k0 landed
    } else {
      asm volatile("s_wait_asynccnt 0x0" ::: "memory");  // drain
    }
    zero_patch(k0, cur);
    __syncthreads();

    AFrag af[4], bfr[2];
    #pragma unroll
    for (int mt = 0; mt < 4; mt++) {
      int r = waveM * 64 + mt * 16 + mIn;
      af[mt].q[0] = *(const uint4*)(&As[cur][r][kB]);
      af[mt].q[1] = *(const uint4*)(&As[cur][r][kB + 16]);
    }
    #pragma unroll
    for (int nt = 0; nt < 2; nt++) {
      int r = waveN * 32 + nt * 16 + mIn;
      bfr[nt].q[0] = *(const uint4*)(&Bs[cur][r][kB]);
      bfr[nt].q[1] = *(const uint4*)(&Bs[cur][r][kB + 16]);
    }
    #pragma unroll
    for (int mt = 0; mt < 4; mt++)
      #pragma unroll
      for (int nt = 0; nt < 2; nt++)
        acc[mt][nt] = __builtin_amdgcn_wmma_f32_16x16x32_bf16(
            false, af[mt].v, false, bfr[nt].v,
            (short)0, acc[mt][nt], false, false);
    __syncthreads();   // all reads done before next tile overwrites buffer
  }

  const int rowHi = (lane < 16) ? 0 : 8;
  #pragma unroll
  for (int mt = 0; mt < 4; mt++) {
    #pragma unroll
    for (int nt = 0; nt < 2; nt++) {
      int gn = blockN0 + waveN * 32 + nt * 16 + mIn;
      float bias = b2[gn];
      #pragma unroll
      for (int r = 0; r < 8; r++) {
        int gm = blockM0 + waveM * 64 + mt * 16 + r + rowHi;
        out[(size_t)gm * 1024 + gn] = acc[mt][nt][r] + bias;
      }
    }
  }
}

// ---------------------------------------------------------------------------
// u_hat[b,caps,jk] = sum_i w[caps,jk,i] * u[b,caps,i]
// u[b,caps,i] = conv2out[(b*256 + h*64 + w16*16 + i)*1024 + co]
//   caps = co*16 + h*4 + w16
// ---------------------------------------------------------------------------
__global__ void uhat_kernel(const float* __restrict__ conv2out,
                            const float* __restrict__ w,
                            float* __restrict__ u_hat) {
  int caps = blockIdx.x;
  int tid = threadIdx.x;
  __shared__ float wsh[1024];
  __shared__ float ush[1024];
  for (int i = tid; i < 1024; i += 256)
    wsh[i] = w[(size_t)caps * 1024 + i];
  int co = caps >> 4, rem = caps & 15, h = rem >> 2, w16 = rem & 3;
  for (int e = tid; e < 1024; e += 256) {
    int b = e >> 4, i = e & 15;
    ush[e] = conv2out[((size_t)((b*4 + h)*64 + (w16*16 + i)))*1024 + co];
  }
  __syncthreads();
  for (int o = tid; o < 4096; o += 256) {
    int b = o >> 6, jk = o & 63;
    float acc = 0.f;
    #pragma unroll
    for (int i = 0; i < 16; i++) acc += wsh[jk*16 + i] * ush[b*16 + i];
    u_hat[((size_t)(b*16384 + caps))*64 + jk] = acc;
  }
}

// ---------------------------------------------------------------------------
// Routing pass 0: s1 partials = 0.5 * sum_n u_hat   (softmax(0) = 0.5)
// ---------------------------------------------------------------------------
__global__ void s_pass0(const float* __restrict__ u_hat,
                        float* __restrict__ partial) {
  int b = blockIdx.x, chunk = blockIdx.y;
  int tid = threadIdx.x, jk = tid & 63, grp = tid >> 6;
  float acc = 0.f;
  for (int nn = grp; nn < 512; nn += 4) {
    int n = chunk * 512 + nn;
    acc += u_hat[((size_t)(b*16384 + n))*64 + jk];
  }
  __shared__ float red[256];
  red[tid] = acc;
  __syncthreads();
  if (tid < 64)
    partial[((size_t)(b*32 + chunk))*64 + tid] =
        0.5f * (red[tid] + red[tid+64] + red[tid+128] + red[tid+192]);
}

// ---------------------------------------------------------------------------
// Fused routing iteration: agreement -> b_log update -> softmax -> s partials
// wave = fixed (grp, j), lanes = k; deterministic (no atomics)
// ---------------------------------------------------------------------------
__global__ void agr_s_pass(const float* __restrict__ u_hat,
                           const float* __restrict__ v,
                           float* __restrict__ b_log,
                           float* __restrict__ partial,
                           int accumulate) {
  int b = blockIdx.x, chunk = blockIdx.y;
  int tid = threadIdx.x;
  int k = tid & 31, j = (tid >> 5) & 1, grp = tid >> 6;
  __shared__ float vsh[64];
  __shared__ float ash[4][2];
  __shared__ float red[256];
  if (tid < 64) vsh[tid] = v[b*64 + tid];
  __syncthreads();
  float vme = vsh[j*32 + k];
  float accs = 0.f;
  int nbase = chunk * 512;
  for (int nn = grp; nn < 512; nn += 4) {
    int n = nbase + nn;
    float uh = u_hat[((size_t)(b*16384 + n))*64 + j*32 + k];
    float a = uh * vme;
    #pragma unroll
    for (int off = 16; off > 0; off >>= 1) a += __shfl_xor(a, off);
    if (k == 0) ash[grp][j] = a;
    __syncthreads();
    size_t bi = (size_t)(b*16384 + n) * 2;
    float n0 = ash[grp][0], n1 = ash[grp][1];
    if (accumulate) { n0 += b_log[bi]; n1 += b_log[bi + 1]; }
    float mx = fmaxf(n0, n1);
    float e0 = __expf(n0 - mx), e1 = __expf(n1 - mx);
    float c = ((j == 0) ? e0 : e1) / (e0 + e1);
    accs += c * uh;
    __syncthreads();                       // all reads of old b_log done
    if (k == 0) b_log[bi + j] = (j == 0) ? n0 : n1;
  }
  red[tid] = accs;
  __syncthreads();
  if (tid < 64) {
    float sum = red[tid] + red[tid+64] + red[tid+128] + red[tid+192];
    partial[((size_t)(b*32 + chunk))*64 + tid] = sum;
  }
}

__global__ void reduce_s(const float* __restrict__ partial,
                         float* __restrict__ s) {
  int idx = blockIdx.x * 256 + threadIdx.x;   // 4096
  int b = idx >> 6, jk = idx & 63;
  float acc = 0.f;
  for (int c = 0; c < 32; c++)
    acc += partial[((size_t)(b*32 + c))*64 + jk];
  s[idx] = acc;
}

// ---------------------------------------------------------------------------
// squash; on final iteration also emits v to d_out and masked v_flat
// ---------------------------------------------------------------------------
__global__ void squash_kernel(const float* __restrict__ s,
                              float* __restrict__ v,
                              float* __restrict__ out_v,
                              float* __restrict__ vflat,
                              const float* __restrict__ y,
                              int final_iter) {
  int bj = blockIdx.x;
  int b = bj >> 1, j = bj & 1, k = threadIdx.x;
  float sv = s[b*64 + j*32 + k];
  float sq = sv * sv;
  #pragma unroll
  for (int off = 16; off > 0; off >>= 1) sq += __shfl_xor(sq, off);
  float norm = sqrtf(sq);
  float val = (sq / (1.f + sq)) * (sv / (norm + 1e-7f));
  v[b*64 + j*32 + k] = val;
  if (final_iter) {
    out_v[b*64 + j*32 + k] = val;
    vflat[b*64 + j*32 + k] = y[b*2 + j] * val;
  }
}

// ---------------------------------------------------------------------------
// Decoder FC: out[m][n] = relu(sum_k A[m][k]*W[k][n] + bias[n]); N % 256 == 0
// ---------------------------------------------------------------------------
__global__ void fc_relu(const float* __restrict__ A,
                        const float* __restrict__ W,
                        const float* __restrict__ bias,
                        float* __restrict__ out, int K, int N) {
  int m = blockIdx.x;
  int n = blockIdx.y * 256 + threadIdx.x;
  float acc = bias[n];
  for (int k = 0; k < K; k++)
    acc += A[(size_t)m*K + k] * W[(size_t)k*N + n];
  out[(size_t)m*N + n] = fmaxf(acc, 0.f);
}

// ---------------------------------------------------------------------------
extern "C" void kernel_launch(void* const* d_in, const int* in_sizes, int n_in,
                              void* d_out, int out_size, void* d_ws,
                              size_t ws_size, hipStream_t stream) {
  (void)in_sizes; (void)n_in; (void)out_size; (void)ws_size;
  const float* x_inp = (const float*)d_in[0];
  const float* y     = (const float*)d_in[1];
  const float* k1    = (const float*)d_in[2];
  const float* b1    = (const float*)d_in[3];
  const float* k2    = (const float*)d_in[4];
  const float* b2    = (const float*)d_in[5];
  const float* w     = (const float*)d_in[6];
  const float* W1    = (const float*)d_in[7];
  const float* d1    = (const float*)d_in[8];
  const float* W2    = (const float*)d_in[9];
  const float* d2    = (const float*)d_in[10];
  const float* W3    = (const float*)d_in[11];
  const float* d3    = (const float*)d_in[12];
  float* out = (float*)d_out;

  char* p = (char*)d_ws;
  __bf16* act1 = (__bf16*)p;  p += (size_t)64*8*128*512*2;   // 67 MB
  __bf16* Wt2  = (__bf16*)p;  p += (size_t)1024*4608*2;      // 9.4 MB
  float*  u    = (float*)p;   p += (size_t)16384*1024*4;     // 67 MB
  float*  uhat = (float*)p;   p += (size_t)64*16384*64*4;    // 268 MB
  float*  blog = (float*)p;   p += (size_t)64*16384*2*4;     // 8 MB
  float*  part = (float*)p;   p += (size_t)64*32*64*4;
  float*  s    = (float*)p;   p += (size_t)4096*4;
  float*  v    = (float*)p;   p += (size_t)4096*4;
  float*  vflt = (float*)p;   p += (size_t)4096*4;
  float*  dec1 = (float*)p;   p += (size_t)64*1024*4;
  float*  dec2 = (float*)p;   p += (size_t)64*2048*4;

  conv1_kernel<<<131072, 256, 0, stream>>>(x_inp, k1, b1, act1);
  wt2_kernel<<<18432, 256, 0, stream>>>(k2, Wt2);
  conv2_gemm<<<dim3(128, 8), 256, 0, stream>>>(act1, Wt2, b2, u);
  uhat_kernel<<<16384, 256, 0, stream>>>(u, w, uhat);

  // routing iteration 1 (c = 0.5)
  s_pass0<<<dim3(64, 32), 256, 0, stream>>>(uhat, part);
  reduce_s<<<16, 256, 0, stream>>>(part, s);
  squash_kernel<<<128, 32, 0, stream>>>(s, v, nullptr, nullptr, y, 0);
  // routing iteration 2
  agr_s_pass<<<dim3(64, 32), 256, 0, stream>>>(uhat, v, blog, part, 0);
  reduce_s<<<16, 256, 0, stream>>>(part, s);
  squash_kernel<<<128, 32, 0, stream>>>(s, v, nullptr, nullptr, y, 0);
  // routing iteration 3 (final v)
  agr_s_pass<<<dim3(64, 32), 256, 0, stream>>>(uhat, v, blog, part, 1);
  reduce_s<<<16, 256, 0, stream>>>(part, s);
  squash_kernel<<<128, 32, 0, stream>>>(s, v, out, vflt, y, 1);

  // decoder
  fc_relu<<<dim3(64, 4), 256, 0, stream>>>(vflt, W1, d1, dec1, 64, 1024);
  fc_relu<<<dim3(64, 8), 256, 0, stream>>>(dec1, W2, d2, dec2, 1024, 2048);
  fc_relu<<<dim3(64, 8), 256, 0, stream>>>(dec2, W3, d3, out + 4096, 2048, 2048);
}
